// LSTM_19421842112761
// MI455X (gfx1250) — matrix operand order repaired and checked by reading, not compiled, not run on hardware
//
#include <hip/hip_runtime.h>
#include <hip/hip_bf16.h>

#define HDIM   2048
#define BDIM   256
#define TSTEPS 128
#define CVOC   64
#define EDIM   1024
#define NKT    (HDIM / 32)   // 64 K-steps per GEMM
#define NMT    (HDIM / 16)   // 128 M-tiles

typedef __attribute__((ext_vector_type(16))) __bf16 v16bf;
typedef __attribute__((ext_vector_type(8)))  __bf16 v8bf;
typedef __attribute__((ext_vector_type(8)))  float  v8f;

// ---------------- workspace layout (bytes) ----------------
// [0,        32 MiB)  Wsw : bf16 swizzled recurrent weights, 4 gates,
//                     fragment-major: (((g*128+mt)*64+kt)*32+lane)*16 elements
// [32 MiB,  +2 MiB)   U   : fp32 [4][64][2048]  (W_*x @ emb^T + b_*)
// [34 MiB,  +1 MiB)   hT0 : bf16 [256][2048]  (h transposed, batch-major)
// [35 MiB,  +1 MiB)   hT1 : bf16 [256][2048]
// [36 MiB]            bar : 2 x u32 grid-barrier state
#define OFF_WSW  0
#define OFF_U    33554432u
#define OFF_HT0  35651584u
#define OFF_HT1  36700160u
#define OFF_BAR  37748736u

// ---------------- fast activations (v_exp_f32 / v_rcp_f32) ----------------
__device__ __forceinline__ float sigm_f(float x) {
    x = fminf(fmaxf(x, -30.f), 30.f);
    return __builtin_amdgcn_rcpf(1.f + __builtin_amdgcn_exp2f(-x * 1.4426950408889634f));
}
__device__ __forceinline__ float tanh_f(float x) {
    x = fminf(fmaxf(x, -15.f), 15.f);
    float e = __builtin_amdgcn_exp2f(-2.885390081777927f * x);   // exp(-2x)
    return (1.f - e) * __builtin_amdgcn_rcpf(1.f + e);
}

// ---------------- kernel 1: U[g][c][h] = W_*x @ emb^T + b ----------------
__global__ __launch_bounds__(256) void build_U(
    const float* __restrict__ Wg, const float* __restrict__ Wi,
    const float* __restrict__ Wf, const float* __restrict__ Wo,
    const float* __restrict__ bg, const float* __restrict__ bi,
    const float* __restrict__ bf_, const float* __restrict__ bo,
    const float* __restrict__ emb, float* __restrict__ U)
{
    int tid = blockIdx.x * blockDim.x + threadIdx.x;   // 4*64*2048 threads
    int h = tid & (HDIM - 1);
    int c = (tid >> 11) & (CVOC - 1);
    int g = tid >> 17;
    const float* W = (g == 0) ? Wg : (g == 1) ? Wi : (g == 2) ? Wf : Wo;
    const float* b = (g == 0) ? bg : (g == 1) ? bi : (g == 2) ? bf_ : bo;
    const float* wrow = W + (size_t)h * EDIM;
    const float* erow = emb + (size_t)c * EDIM;
    float s = b[h];
    for (int e = 0; e < EDIM; ++e) s += wrow[e] * erow[e];
    U[tid] = s;   // tid == (g*64 + c)*2048 + h
}

// ---------------- kernel 2: fp32 -> bf16, A-fragment swizzle ----------------
// Per ISA 16-bit A 16x32 layout: lanes 0-15 row M=lane, K {0..7,16..23};
// lanes 16-31 row M=lane-16, K {8..15,24..31}.
__global__ __launch_bounds__(256) void swizzle_W(
    const float* __restrict__ Wg, const float* __restrict__ Wi,
    const float* __restrict__ Wf, const float* __restrict__ Wo,
    __bf16* __restrict__ Wsw)
{
    int tid  = blockIdx.x * blockDim.x + threadIdx.x;   // 2^20 threads
    int lane = tid & 31;
    int kt   = (tid >> 5) & (NKT - 1);
    int mt   = (tid >> 11) & (NMT - 1);
    int g    = tid >> 18;
    const float* W = (g == 0) ? Wg : (g == 1) ? Wi : (g == 2) ? Wf : Wo;
    int m    = mt * 16 + (lane & 15);
    int half = (lane >> 4) << 3;                        // 0 or 8
    const float* row = W + (size_t)m * HDIM + (size_t)kt * 32;
    v16bf out;
    #pragma unroll
    for (int j = 0; j < 8; ++j) out[j]     = (__bf16)row[half + j];
    #pragma unroll
    for (int j = 0; j < 8; ++j) out[8 + j] = (__bf16)row[16 + half + j];
    ((v16bf*)Wsw)[(((size_t)g * NMT + mt) * NKT + kt) * 32 + lane] = out;
}

// ---------------- kernel 3: zero h buffers + barrier ----------------
__global__ __launch_bounds__(256) void init_state(
    __bf16* __restrict__ hT0, __bf16* __restrict__ hT1, unsigned* __restrict__ bar)
{
    int tid = blockIdx.x * blockDim.x + threadIdx.x;    // 256*2048 threads
    hT0[tid] = (__bf16)0.f;
    hT1[tid] = (__bf16)0.f;
    if (tid < 2) bar[tid] = 0u;
}

// ---------------- kernel 4: persistent LSTM recurrence ----------------
__global__ __launch_bounds__(128, 1) void lstm_persistent(
    const __bf16* __restrict__ Wsw, const float* __restrict__ U,
    __bf16* __restrict__ hT0, __bf16* __restrict__ hT1,
    const int* __restrict__ xidx, unsigned* __restrict__ bar)
{
    const int lane   = threadIdx.x & 31;
    const int wave   = threadIdx.x >> 5;        // 0..3
    const int ng     = blockIdx.x & 3;          // batch group (64 cols)
    const int mg     = blockIdx.x >> 2;         // 0..31
    const int mt     = mg * 4 + wave;           // 0..127 : 16-row M tile
    const int l15    = lane & 15;
    const int half8  = (lane >> 4) << 3;        // D-frag rows m0..m0+7
    const int m0     = mt * 16 + half8;
    const int nbase  = ng * 64;
    const int hoff16 = (lane >> 4) << 3;        // B-frag K offset (0|8)
    const unsigned nblocks = gridDim.x;

    // per-gate A-fragment lane pointers (advance 32 v16bf per kt)
    const v16bf* wp[4];
    #pragma unroll
    for (int g = 0; g < 4; ++g)
        wp[g] = (const v16bf*)Wsw + ((size_t)g * NMT + mt) * NKT * 32 + lane;

    float cst[4][8];                            // cell state lives in VGPRs
    #pragma unroll
    for (int nt = 0; nt < 4; ++nt)
        #pragma unroll
        for (int r = 0; r < 8; ++r) cst[nt][r] = 0.f;

    for (int t = 0; t < TSTEPS; ++t) {
        const __bf16* __restrict__ hprev = (t & 1) ? hT1 : hT0;
        __bf16*       __restrict__ hnext = (t & 1) ? hT0 : hT1;

        v8f acc[4][4];
        v8f zz = {};
        #pragma unroll
        for (int g = 0; g < 4; ++g)
            #pragma unroll
            for (int nt = 0; nt < 4; ++nt) acc[g][nt] = zz;

        const __bf16* hp[4];
        #pragma unroll
        for (int nt = 0; nt < 4; ++nt)
            hp[nt] = hprev + (size_t)(nbase + nt * 16 + l15) * HDIM + hoff16;

        for (int kt = 0; kt < NKT; ++kt) {
            v16bf afr[4], bfr[4];
            #pragma unroll
            for (int g = 0; g < 4; ++g) afr[g] = wp[g][(size_t)kt * 32];
            #pragma unroll
            for (int nt = 0; nt < 4; ++nt) {
                v8bf lo = *(const v8bf*)(hp[nt] + (size_t)kt * 32);
                v8bf hi = *(const v8bf*)(hp[nt] + (size_t)kt * 32 + 16);
                bfr[nt] = __builtin_shufflevector(lo, hi, 0,1,2,3,4,5,6,7,8,9,10,11,12,13,14,15);
            }
            #pragma unroll
            for (int g = 0; g < 4; ++g)
                #pragma unroll
                for (int nt = 0; nt < 4; ++nt)
                    acc[g][nt] = __builtin_amdgcn_wmma_f32_16x16x32_bf16(
                        false, afr[g], false, bfr[nt], (short)0, acc[g][nt],
                        false, false);
        }

        // epilogue: add vocab-lookup input term, activations, c/h update
        #pragma unroll
        for (int nt = 0; nt < 4; ++nt) {
            const int n   = nbase + nt * 16 + l15;
            const int idx = xidx[n * TSTEPS + t];
            v8f vg = *(const v8f*)(U + ((size_t)(0 * CVOC + idx)) * HDIM + m0);
            v8f vi = *(const v8f*)(U + ((size_t)(1 * CVOC + idx)) * HDIM + m0);
            v8f vf = *(const v8f*)(U + ((size_t)(2 * CVOC + idx)) * HDIM + m0);
            v8f vo = *(const v8f*)(U + ((size_t)(3 * CVOC + idx)) * HDIM + m0);
            v8bf hv;
            #pragma unroll
            for (int r = 0; r < 8; ++r) {
                float gg = tanh_f(acc[0][nt][r] + vg[r]);
                float ii = sigm_f(acc[1][nt][r] + vi[r]);
                float ff = sigm_f(acc[2][nt][r] + vf[r]);
                float oo = sigm_f(acc[3][nt][r] + vo[r]);
                float cn = gg * ii + cst[nt][r] * ff;
                cst[nt][r] = cn;
                hv[r] = (__bf16)(tanh_f(cn) * oo);
            }
            *(v8bf*)(hnext + (size_t)n * HDIM + m0) = hv;   // 16B store
        }

        // ---- device-wide barrier between timesteps ----
        __threadfence();
        __syncthreads();
        if (threadIdx.x == 0) {
            unsigned gv = __hip_atomic_load(&bar[1], __ATOMIC_RELAXED,
                                            __HIP_MEMORY_SCOPE_AGENT);
            unsigned a  = __hip_atomic_fetch_add(&bar[0], 1u, __ATOMIC_ACQ_REL,
                                                 __HIP_MEMORY_SCOPE_AGENT);
            if (a == nblocks - 1) {
                __hip_atomic_store(&bar[0], 0u, __ATOMIC_RELAXED,
                                   __HIP_MEMORY_SCOPE_AGENT);
                __hip_atomic_fetch_add(&bar[1], 1u, __ATOMIC_RELEASE,
                                       __HIP_MEMORY_SCOPE_AGENT);
            } else {
                while (__hip_atomic_load(&bar[1], __ATOMIC_ACQUIRE,
                                         __HIP_MEMORY_SCOPE_AGENT) == gv)
                    __builtin_amdgcn_s_sleep(1);
            }
        }
        __syncthreads();
    }
}

// ---------------- kernel 5: p = W_ph @ h + b_p ; log_softmax ----------------
__global__ __launch_bounds__(64) void project_softmax(
    const __bf16* __restrict__ hT, const float* __restrict__ Wph,
    const float* __restrict__ bp, float* __restrict__ y)
{
    __shared__ float p[CVOC];
    const int b = blockIdx.x;          // batch column
    const int c = threadIdx.x;         // vocab row
    const __bf16* hcol = hT  + (size_t)b * HDIM;
    const float*  wrow = Wph + (size_t)c * HDIM;
    float s = bp[c];
    for (int k = 0; k < HDIM; ++k) s += wrow[k] * (float)hcol[k];
    p[c] = s;
    __syncthreads();
    float m = -3.0e38f;
    for (int j = 0; j < CVOC; ++j) m = fmaxf(m, p[j]);
    float se = 0.f;
    for (int j = 0; j < CVOC; ++j)
        se += __builtin_amdgcn_exp2f((p[j] - m) * 1.4426950408889634f);
    y[(size_t)b * CVOC + c] = s - m - logf(se);
}

extern "C" void kernel_launch(void* const* d_in, const int* in_sizes, int n_in,
                              void* d_out, int out_size, void* d_ws, size_t ws_size,
                              hipStream_t stream) {
    const int*   x   = (const int*)  d_in[0];
    const float* emb = (const float*)d_in[1];
    const float* Wgx = (const float*)d_in[2];
    const float* Wix = (const float*)d_in[3];
    const float* Wfx = (const float*)d_in[4];
    const float* Wox = (const float*)d_in[5];
    const float* Wgh = (const float*)d_in[6];
    const float* Wih = (const float*)d_in[7];
    const float* Wfh = (const float*)d_in[8];
    const float* Woh = (const float*)d_in[9];
    const float* bg  = (const float*)d_in[10];
    const float* bi  = (const float*)d_in[11];
    const float* bf_ = (const float*)d_in[12];
    const float* bo  = (const float*)d_in[13];
    const float* Wph = (const float*)d_in[14];
    const float* bp  = (const float*)d_in[15];
    float* y = (float*)d_out;

    char* ws = (char*)d_ws;
    __bf16*   Wsw = (__bf16*)(ws + OFF_WSW);
    float*    U   = (float*) (ws + OFF_U);
    __bf16*   hT0 = (__bf16*)(ws + OFF_HT0);
    __bf16*   hT1 = (__bf16*)(ws + OFF_HT1);
    unsigned* bar = (unsigned*)(ws + OFF_BAR);

    build_U<<<2048, 256, 0, stream>>>(Wgx, Wix, Wfx, Wox, bg, bi, bf_, bo, emb, U);
    swizzle_W<<<4096, 256, 0, stream>>>(Wgh, Wih, Wfh, Woh, Wsw);
    init_state<<<2048, 256, 0, stream>>>(hT0, hT1, bar);
    lstm_persistent<<<128, 128, 0, stream>>>(Wsw, U, hT0, hT1, x, bar);
    // T even -> final h lives in hT0
    project_softmax<<<BDIM, CVOC, 0, stream>>>(hT0, Wph, bp, y);
}
